// SplineConvNet_11940009083383
// MI455X (gfx1250) — matrix operand order repaired and checked by reading, not compiled, not run on hardware
//
#include <hip/hip_runtime.h>

// ---------------- types for CDNA5 WMMA ----------------
typedef __attribute__((ext_vector_type(16))) __bf16        v16bf;
typedef __attribute__((ext_vector_type(8)))  float         v8f;
typedef __attribute__((ext_vector_type(4)))  unsigned int  v4u;

#define N_NODES 40000
#define N_EDGES 640000
#define F_IN    6
#define HIDC    128
#define KTOT    27          // 3^3 kernel weight matrices
#define NGRAPH  8
#define CLS     10
#define K1P     192         // 27*6 = 162 bucket cols + 6 root cols, padded to mult of 32
#define K2P     3584        // 27*128 = 3456 bucket cols + 128 root cols
#define K3P     256         // MLP input (concat x1,x2)
#define BN_EPS  1e-5f

static __device__ __forceinline__ unsigned short f2bf(float f) {
  unsigned u = __builtin_bit_cast(unsigned, f);
  u += 0x7FFFu + ((u >> 16) & 1u);          // round-to-nearest-even
  return (unsigned short)(u >> 16);
}
static __device__ __forceinline__ float bf2f(unsigned short h) {
  return __builtin_bit_cast(float, ((unsigned)h) << 16);
}

// ---------------- edge preparation: spline basis + kernel index + degree ----
__global__ void k_edge_prep(const float* __restrict__ pseudo,
                            const int* __restrict__ dst,
                            float* __restrict__ basis, int* __restrict__ widx,
                            float* __restrict__ deg) {
  int e = blockIdx.x * blockDim.x + threadIdx.x;
  if (e >= N_EDGES) return;
  float fr[3]; int lo[3];
#pragma unroll
  for (int d = 0; d < 3; ++d) {
    float v = pseudo[e * 3 + d] * 2.0f;     // pseudo * (K-1)
    float fl = floorf(v);
    fr[d] = v - fl;
    lo[d] = (int)fl;
  }
#pragma unroll
  for (int s = 0; s < 8; ++s) {
    float b = 1.0f; int wi = 0; int pw = 1;
#pragma unroll
    for (int d = 0; d < 3; ++d) {
      int bit = (s >> d) & 1;
      b *= fabsf(1.0f - fr[d] - (float)bit);
      wi += ((lo[d] + bit) % 3) * pw;
      pw *= 3;
    }
    basis[e * 8 + s] = b;
    widx[e * 8 + s] = wi;
  }
  atomicAdd(&deg[dst[e]], 1.0f);
}

// ---------------- layer-1 scatter: A1f[dst, wi*6 + f] += basis * x[src, f] --
__global__ void k_scatter1(const float* __restrict__ x,
                           const int* __restrict__ src, const int* __restrict__ dst,
                           const float* __restrict__ basis, const int* __restrict__ widx,
                           float* __restrict__ A1f) {
  int e = blockIdx.x * blockDim.x + threadIdx.x;
  if (e >= N_EDGES) return;
  int s_ = src[e], d_ = dst[e];
  float xv[F_IN];
#pragma unroll
  for (int f = 0; f < F_IN; ++f) xv[f] = x[s_ * F_IN + f];
#pragma unroll
  for (int s = 0; s < 8; ++s) {
    float b = basis[e * 8 + s];
    int base = d_ * K1P + widx[e * 8 + s] * F_IN;
#pragma unroll
    for (int f = 0; f < F_IN; ++f)
      atomicAdd(&A1f[base + f], b * xv[f]);
  }
}

// ---------------- layer-2 scatter: 128 channels per edge, from bf16 x1 ------
__global__ void k_scatter2(const unsigned short* __restrict__ xcat,
                           const int* __restrict__ src, const int* __restrict__ dst,
                           const float* __restrict__ basis, const int* __restrict__ widx,
                           float* __restrict__ A2f) {
  long long gid = (long long)blockIdx.x * blockDim.x + threadIdx.x;
  if (gid >= (long long)N_EDGES * HIDC) return;
  int e = (int)(gid >> 7);
  int t = (int)(gid & 127);
  int s_ = src[e], d_ = dst[e];
  float xv = bf2f(xcat[(size_t)s_ * 256 + t]);   // x1 lives in cols [0,128)
  size_t rowbase = (size_t)d_ * (KTOT * HIDC);
#pragma unroll
  for (int s = 0; s < 8; ++s) {
    float b = basis[e * 8 + s];
    int wi = widx[e * 8 + s];
    atomicAdd(&A2f[rowbase + wi * HIDC + t], b * xv);
  }
}

// ---- build bf16 GEMM-A for layer1: buckets/deg | x (root cols) | zero pad --
__global__ void k_convert1(const float* __restrict__ A1f, const float* __restrict__ x,
                           const float* __restrict__ deg, unsigned short* __restrict__ A1h) {
  int i = blockIdx.x * blockDim.x + threadIdx.x;
  if (i >= N_NODES * K1P) return;
  int n = i / K1P, c = i % K1P;
  float v;
  if (c < KTOT * F_IN) {
    float dg = deg[n]; dg = dg > 1.0f ? dg : 1.0f;
    v = A1f[i] / dg;
  } else if (c < KTOT * F_IN + F_IN) {
    v = x[n * F_IN + (c - KTOT * F_IN)];
  } else {
    v = 0.0f;
  }
  A1h[i] = f2bf(v);
}

// ---- build bf16 GEMM-A for layer2: buckets/deg | x1 (root cols) -----------
__global__ void k_convert2(const float* __restrict__ A2f, const unsigned short* __restrict__ xcat,
                           const float* __restrict__ deg, unsigned short* __restrict__ A2h) {
  long long i = (long long)blockIdx.x * blockDim.x + threadIdx.x;
  if (i >= (long long)N_NODES * K2P) return;
  int n = (int)(i / K2P), c = (int)(i % K2P);
  if (c < KTOT * HIDC) {
    float dg = deg[n]; dg = dg > 1.0f ? dg : 1.0f;
    A2h[i] = f2bf(A2f[(size_t)n * (KTOT * HIDC) + c] / dg);
  } else {
    A2h[i] = xcat[(size_t)n * 256 + (c - KTOT * HIDC)];
  }
}

// ---------------- row-major bf16 weight assembly ---------------------------
__global__ void k_build_w1(const float* __restrict__ W1, const float* __restrict__ root1,
                           unsigned short* __restrict__ Wrow) {
  int i = blockIdx.x * blockDim.x + threadIdx.x;
  if (i >= K1P * HIDC) return;
  int r = i / HIDC, c = i % HIDC;
  float v = 0.0f;
  if (r < KTOT * F_IN)            v = W1[(r / F_IN) * (F_IN * HIDC) + (r % F_IN) * HIDC + c];
  else if (r < KTOT * F_IN + F_IN) v = root1[(r - KTOT * F_IN) * HIDC + c];
  Wrow[i] = f2bf(v);
}
__global__ void k_build_w2(const float* __restrict__ W2, const float* __restrict__ root2,
                           unsigned short* __restrict__ Wrow) {
  int i = blockIdx.x * blockDim.x + threadIdx.x;
  if (i >= K2P * HIDC) return;
  int r = i / HIDC, c = i % HIDC;
  float v = (r < KTOT * HIDC) ? W2[(r / HIDC) * (HIDC * HIDC) + (r % HIDC) * HIDC + c]
                              : root2[(r - KTOT * HIDC) * HIDC + c];
  Wrow[i] = f2bf(v);
}
__global__ void k_build_w3(const float* __restrict__ mlp_w, unsigned short* __restrict__ Wrow) {
  int i = blockIdx.x * blockDim.x + threadIdx.x;
  if (i >= 256 * 256) return;
  Wrow[i] = f2bf(mlp_w[i]);
}

// ---- pack row-major [K,N] bf16 into staged WMMA-B layout ------------------
// 16-byte chunks ordered [kt][nt][half][lane]; per-lane v16bf element e maps
// to K = kt*32 + half*16 + (lane>>4)*8 + j8, N = nt*16 + (lane&15), e = half*8 + j8.
__global__ void k_pack_b(const unsigned short* __restrict__ Wrow, unsigned short* __restrict__ Wp,
                         int Kp, int Nout) {
  int NTn = Nout >> 4;
  long long i = (long long)blockIdx.x * blockDim.x + threadIdx.x;
  if (i >= (long long)Kp * Nout) return;
  int j8   = (int)(i & 7);
  long long c = i >> 3;                 // 16B chunk index
  int lane = (int)(c & 31);
  int h    = (int)((c >> 5) & 1);
  long long rest = c >> 6;
  int nt = (int)(rest % NTn);
  int kt = (int)(rest / NTn);
  int k = kt * 32 + h * 16 + (lane >> 4) * 8 + j8;
  int n = nt * 16 + (lane & 15);
  Wp[i] = Wrow[(size_t)k * Nout + n];
}

// ---------------- WMMA GEMM: Y = relu(A @ W + bias), + BN batch stats ------
// 128 threads = 4 waves/block, one 16-row M-tile per wave; B tiles double-
// buffered through LDS (shared by the 4 waves -> 4x less L2 B-traffic and
// the WMMA operand dependency becomes ds_load instead of an L2 round trip).
template <int NT>
__global__ __launch_bounds__(128) void k_gemm_wmma(
    const unsigned short* __restrict__ A,   // [M, Kdim] bf16 row-major
    const unsigned short* __restrict__ Wp,  // packed B chunks
    const float* __restrict__ bias,         // [NT*16]
    float* __restrict__ Y,                  // [M, NT*16]
    float* __restrict__ stats,              // [2*NT*16] sum|sumsq
    int M, int Kdim) {
  __shared__ v4u smem[2 * NT * 64];         // two kt-stage buffers
  const int tid  = threadIdx.x;
  const int lane = tid & 31;
  const int mtile = blockIdx.x * 4 + (tid >> 5);
  const int hv = lane >> 4;
  const unsigned short* pA = A + (size_t)(mtile * 16 + (lane & 15)) * Kdim + hv * 8;
  const v4u* wp4 = (const v4u*)Wp;

  v8f acc[NT];
  v8f zero = {0.f, 0.f, 0.f, 0.f, 0.f, 0.f, 0.f, 0.f};
#pragma unroll
  for (int nt = 0; nt < NT; ++nt) acc[nt] = zero;

  const int ktn = Kdim >> 5;
  v4u stage[NT / 2];
  // preload B(kt=0) -> LDS buffer 0 (identity chunk mapping: coalesced + no bank conflicts)
#pragma unroll
  for (int i = 0; i < NT / 2; ++i) stage[i] = wp4[tid + 128 * i];
#pragma unroll
  for (int i = 0; i < NT / 2; ++i) smem[tid + 128 * i] = stage[i];
  __syncthreads();

  for (int kt = 0; kt < ktn; ++kt) {
    const int cur = (kt & 1) * (NT * 64);
    if (kt + 1 < ktn) {
#pragma unroll
      for (int i = 0; i < NT / 2; ++i)
        stage[i] = wp4[(size_t)(kt + 1) * (NT * 64) + tid + 128 * i];
    }
    // A tile: two 16B chunks per lane (ISA 16-bit A layout)
    v4u c0 = *(const v4u*)(pA + kt * 32);
    v4u c1 = *(const v4u*)(pA + kt * 32 + 16);
    v16bf a = __builtin_bit_cast(v16bf, __builtin_shufflevector(c0, c1, 0, 1, 2, 3, 4, 5, 6, 7));
    __builtin_prefetch(pA + kt * 32 + 64, 0, 3);   // A tile two k-steps ahead (kt+2), near caches
#pragma unroll
    for (int nt = 0; nt < NT; ++nt) {
      v4u lo = smem[cur + nt * 64 + lane];
      v4u hi = smem[cur + nt * 64 + 32 + lane];
      v16bf b = __builtin_bit_cast(v16bf, __builtin_shufflevector(lo, hi, 0, 1, 2, 3, 4, 5, 6, 7));
      acc[nt] = __builtin_amdgcn_wmma_f32_16x16x32_bf16(
          false, a, false, b, (short)0, acc[nt], false, false);
    }
    if (kt + 1 < ktn) {
      __syncthreads();                      // everyone done reading other buffer
      const int nxt = ((kt + 1) & 1) * (NT * 64);
#pragma unroll
      for (int i = 0; i < NT / 2; ++i) smem[nxt + tid + 128 * i] = stage[i];
      __syncthreads();                      // stores visible before next reads
    }
  }

  // epilogue: bias + relu + store + per-channel BN statistics
  int coln = lane & 15;
#pragma unroll
  for (int nt = 0; nt < NT; ++nt) {
    int col = nt * 16 + coln;
    float bv = bias[col];
    float s = 0.f, s2 = 0.f;
#pragma unroll
    for (int r = 0; r < 8; ++r) {
      float v = acc[nt][r] + bv;
      v = v > 0.f ? v : 0.f;
      Y[(size_t)(mtile * 16 + hv * 8 + r) * (NT * 16) + col] = v;
      s += v; s2 += v * v;
    }
    atomicAdd(&stats[col], s);
    atomicAdd(&stats[NT * 16 + col], s2);
  }
}

// ---------------- BatchNorm coefficients from batch stats ------------------
__global__ void k_bn_coeff(const float* __restrict__ stats, const float* __restrict__ g,
                           const float* __restrict__ b, float invn, int C,
                           float* __restrict__ coeff) {
  int c = blockIdx.x * blockDim.x + threadIdx.x;
  if (c >= C) return;
  float mean = stats[c] * invn;
  float var  = stats[C + c] * invn - mean * mean;   // biased variance
  float sc = g[c] * rsqrtf(var + BN_EPS);
  coeff[c]     = sc;
  coeff[C + c] = b[c] - mean * sc;
}

// apply BN and write bf16 into xcat column block (stride 256)
__global__ void k_bn_to_xcat(const float* __restrict__ Y, const float* __restrict__ coeff,
                             unsigned short* __restrict__ xcat, int coloff) {
  int i = blockIdx.x * blockDim.x + threadIdx.x;
  if (i >= N_NODES * HIDC) return;
  int n = i / HIDC, c = i % HIDC;
  float v = Y[i] * coeff[c] + coeff[HIDC + c];
  xcat[(size_t)n * 256 + coloff + c] = f2bf(v);
}

// apply BN to MLP output and atomic-max into per-graph pool
__global__ void k_bn_max(const float* __restrict__ Y, const float* __restrict__ coeff,
                         const int* __restrict__ batch, float* __restrict__ gmax) {
  int i = blockIdx.x * blockDim.x + threadIdx.x;
  if (i >= N_NODES * 256) return;
  int n = i >> 8, c = i & 255;
  float v = Y[i] * coeff[c] + coeff[256 + c];
  float* addr = &gmax[batch[n] * 256 + c];
  if (v >= 0.f) atomicMax((int*)addr, __float_as_int(v));
  else          atomicMin((unsigned int*)addr, __float_as_uint(v));
}

__global__ void k_init_gmax(float* __restrict__ gmax) {
  int i = blockIdx.x * blockDim.x + threadIdx.x;
  if (i < NGRAPH * 256) gmax[i] = -INFINITY;
}

// ---------------- tiny head: fc1+bn, fc2+bn, fc3, log_softmax (one block) --
__global__ void k_head(const float* __restrict__ gmax,
                       const float* __restrict__ f1w, const float* __restrict__ f1b,
                       const float* __restrict__ f1g, const float* __restrict__ f1be,
                       const float* __restrict__ f2w, const float* __restrict__ f2b,
                       const float* __restrict__ f2g, const float* __restrict__ f2be,
                       const float* __restrict__ f3w, const float* __restrict__ f3b,
                       float* __restrict__ out) {
  __shared__ float sg[NGRAPH][256];
  __shared__ float sz[NGRAPH][256];
  __shared__ float z3[NGRAPH * CLS];
  int t = threadIdx.x;
  for (int i = t; i < NGRAPH * 256; i += 256) sg[i >> 8][i & 255] = gmax[i];
  __syncthreads();
  { // fc1: [8,256] @ [256,256]
    float acc[NGRAPH];
#pragma unroll
    for (int r = 0; r < NGRAPH; ++r) acc[r] = 0.f;
    for (int k = 0; k < 256; ++k) {
      float w = f1w[k * 256 + t];
#pragma unroll
      for (int r = 0; r < NGRAPH; ++r) acc[r] += sg[r][k] * w;
    }
#pragma unroll
    for (int r = 0; r < NGRAPH; ++r) {
      float v = acc[r] + f1b[t];
      sz[r][t] = v > 0.f ? v : 0.f;
    }
  }
  __syncthreads();
  { // bn over 8 rows
    float m = 0.f;
    for (int r = 0; r < NGRAPH; ++r) m += sz[r][t];
    m *= (1.0f / NGRAPH);
    float v2 = 0.f;
    for (int r = 0; r < NGRAPH; ++r) { float d = sz[r][t] - m; v2 += d * d; }
    v2 *= (1.0f / NGRAPH);
    float sc = f1g[t] * rsqrtf(v2 + BN_EPS), sh = f1be[t] - m * sc;
    for (int r = 0; r < NGRAPH; ++r) sg[r][t] = sz[r][t] * sc + sh;
  }
  __syncthreads();
  if (t < 128) { // fc2: [8,256] @ [256,128]
    float acc[NGRAPH];
#pragma unroll
    for (int r = 0; r < NGRAPH; ++r) acc[r] = 0.f;
    for (int k = 0; k < 256; ++k) {
      float w = f2w[k * 128 + t];
#pragma unroll
      for (int r = 0; r < NGRAPH; ++r) acc[r] += sg[r][k] * w;
    }
#pragma unroll
    for (int r = 0; r < NGRAPH; ++r) {
      float v = acc[r] + f2b[t];
      sz[r][t] = v > 0.f ? v : 0.f;
    }
  }
  __syncthreads();
  if (t < 128) { // bn2
    float m = 0.f;
    for (int r = 0; r < NGRAPH; ++r) m += sz[r][t];
    m *= (1.0f / NGRAPH);
    float v2 = 0.f;
    for (int r = 0; r < NGRAPH; ++r) { float d = sz[r][t] - m; v2 += d * d; }
    v2 *= (1.0f / NGRAPH);
    float sc = f2g[t] * rsqrtf(v2 + BN_EPS), sh = f2be[t] - m * sc;
    for (int r = 0; r < NGRAPH; ++r) sg[r][t] = sz[r][t] * sc + sh;
  }
  __syncthreads();
  if (t < NGRAPH * CLS) { // fc3: [8,128] @ [128,10]
    int r = t / CLS, c = t % CLS;
    float a = f3b[c];
    for (int k = 0; k < 128; ++k) a += sg[r][k] * f3w[k * CLS + c];
    z3[t] = a;
  }
  __syncthreads();
  if (t < NGRAPH) { // log_softmax per row
    float m = -INFINITY;
    for (int c = 0; c < CLS; ++c) m = fmaxf(m, z3[t * CLS + c]);
    float s = 0.f;
    for (int c = 0; c < CLS; ++c) s += expf(z3[t * CLS + c] - m);
    float l = logf(s);
    for (int c = 0; c < CLS; ++c) out[t * CLS + c] = z3[t * CLS + c] - m - l;
  }
}

// ---------------- host-side orchestration ----------------------------------
static inline unsigned gdiv(long long n, int b) { return (unsigned)((n + b - 1) / b); }

extern "C" void kernel_launch(void* const* d_in, const int* in_sizes, int n_in,
                              void* d_out, int out_size, void* d_ws, size_t ws_size,
                              hipStream_t stream) {
  (void)in_sizes; (void)n_in; (void)out_size; (void)ws_size;
  const float* x      = (const float*)d_in[0];
  const float* pseudo = (const float*)d_in[1];
  const float* W1     = (const float*)d_in[2];
  const float* root1  = (const float*)d_in[3];
  const float* bias1  = (const float*)d_in[4];
  const float* bn1_g  = (const float*)d_in[5];
  const float* bn1_b  = (const float*)d_in[6];
  const float* W2     = (const float*)d_in[7];
  const float* root2  = (const float*)d_in[8];
  const float* bias2  = (const float*)d_in[9];
  const float* bn2_g  = (const float*)d_in[10];
  const float* bn2_b  = (const float*)d_in[11];
  const float* mlp_w  = (const float*)d_in[12];
  const float* mlp_b  = (const float*)d_in[13];
  const float* mlp_g  = (const float*)d_in[14];
  const float* mlp_be = (const float*)d_in[15];
  const float* f1w    = (const float*)d_in[16];
  const float* f1b    = (const float*)d_in[17];
  const float* f1g    = (const float*)d_in[18];
  const float* f1be   = (const float*)d_in[19];
  const float* f2w    = (const float*)d_in[20];
  const float* f2b    = (const float*)d_in[21];
  const float* f2g    = (const float*)d_in[22];
  const float* f2be   = (const float*)d_in[23];
  const float* f3w    = (const float*)d_in[24];
  const float* f3b    = (const float*)d_in[25];
  const int* edge_index = (const int*)d_in[26];
  const int* batch      = (const int*)d_in[27];
  const int* src = edge_index;
  const int* dst = edge_index + N_EDGES;

  char* ws = (char*)d_ws;
  size_t off = 0;
  auto alloc = [&](size_t bytes) -> char* {
    char* p = ws + off;
    off = (off + bytes + 255) & ~(size_t)255;
    return p;
  };
  float*          basisB = (float*)alloc((size_t)N_EDGES * 8 * 4);
  int*            widxB  = (int*)alloc((size_t)N_EDGES * 8 * 4);
  float*          deg    = (float*)alloc((size_t)N_NODES * 4);
  float*          A1f    = (float*)alloc((size_t)N_NODES * K1P * 4);
  unsigned short* A1h    = (unsigned short*)alloc((size_t)N_NODES * K1P * 2);
  unsigned short* Wrow1  = (unsigned short*)alloc((size_t)K1P * HIDC * 2);
  unsigned short* W1p    = (unsigned short*)alloc((size_t)K1P * HIDC * 2);
  float*          y1     = (float*)alloc((size_t)N_NODES * HIDC * 4);
  float*          stats1 = (float*)alloc(2 * HIDC * 4);
  float*          coeff1 = (float*)alloc(2 * HIDC * 4);
  unsigned short* xcat   = (unsigned short*)alloc((size_t)N_NODES * 256 * 2);
  float*          A2f    = (float*)alloc((size_t)N_NODES * (KTOT * HIDC) * 4);
  unsigned short* A2h    = (unsigned short*)alloc((size_t)N_NODES * K2P * 2);
  unsigned short* Wrow2  = (unsigned short*)alloc((size_t)K2P * HIDC * 2);
  unsigned short* W2p    = (unsigned short*)alloc((size_t)K2P * HIDC * 2);
  float*          y2     = (float*)alloc((size_t)N_NODES * HIDC * 4);
  float*          stats2 = (float*)alloc(2 * HIDC * 4);
  float*          coeff2 = (float*)alloc(2 * HIDC * 4);
  unsigned short* Wrow3  = (unsigned short*)alloc((size_t)256 * 256 * 2);
  unsigned short* W3p    = (unsigned short*)alloc((size_t)256 * 256 * 2);
  float*          y3     = (float*)alloc((size_t)N_NODES * 256 * 4);
  float*          stats3 = (float*)alloc(2 * 256 * 4);
  float*          coeff3 = (float*)alloc(2 * 256 * 4);
  float*          gmax   = (float*)alloc((size_t)NGRAPH * 256 * 4);

  // zero accumulators (graph-capture safe)
  hipMemsetAsync(deg,    0, (size_t)N_NODES * 4, stream);
  hipMemsetAsync(A1f,    0, (size_t)N_NODES * K1P * 4, stream);
  hipMemsetAsync(A2f,    0, (size_t)N_NODES * (KTOT * HIDC) * 4, stream);
  hipMemsetAsync(stats1, 0, 2 * HIDC * 4, stream);
  hipMemsetAsync(stats2, 0, 2 * HIDC * 4, stream);
  hipMemsetAsync(stats3, 0, 2 * 256 * 4, stream);
  k_init_gmax<<<gdiv(NGRAPH * 256, 256), 256, 0, stream>>>(gmax);

  // weight assembly + WMMA-B packing (cheap; L2-resident at run time)
  k_build_w1<<<gdiv((long long)K1P * HIDC, 256), 256, 0, stream>>>(W1, root1, Wrow1);
  k_pack_b<<<gdiv((long long)K1P * HIDC, 256), 256, 0, stream>>>(Wrow1, W1p, K1P, HIDC);
  k_build_w2<<<gdiv((long long)K2P * HIDC, 256), 256, 0, stream>>>(W2, root2, Wrow2);
  k_pack_b<<<gdiv((long long)K2P * HIDC, 256), 256, 0, stream>>>(Wrow2, W2p, K2P, HIDC);
  k_build_w3<<<gdiv(256 * 256, 256), 256, 0, stream>>>(mlp_w, Wrow3);
  k_pack_b<<<gdiv(256 * 256, 256), 256, 0, stream>>>(Wrow3, W3p, K3P, 256);

  // edge prep + layer 1
  k_edge_prep<<<gdiv(N_EDGES, 256), 256, 0, stream>>>(pseudo, dst, basisB, widxB, deg);
  k_scatter1<<<gdiv(N_EDGES, 256), 256, 0, stream>>>(x, src, dst, basisB, widxB, A1f);
  k_convert1<<<gdiv((long long)N_NODES * K1P, 256), 256, 0, stream>>>(A1f, x, deg, A1h);
  k_gemm_wmma<8><<<N_NODES / 64, 128, 0, stream>>>(A1h, W1p, bias1, y1, stats1, N_NODES, K1P);
  k_bn_coeff<<<1, HIDC, 0, stream>>>(stats1, bn1_g, bn1_b, 1.0f / N_NODES, HIDC, coeff1);
  k_bn_to_xcat<<<gdiv((long long)N_NODES * HIDC, 256), 256, 0, stream>>>(y1, coeff1, xcat, 0);

  // layer 2
  k_scatter2<<<gdiv((long long)N_EDGES * HIDC, 256), 256, 0, stream>>>(xcat, src, dst, basisB, widxB, A2f);
  k_convert2<<<gdiv((long long)N_NODES * K2P, 256), 256, 0, stream>>>(A2f, xcat, deg, A2h);
  k_gemm_wmma<8><<<N_NODES / 64, 128, 0, stream>>>(A2h, W2p, bias2, y2, stats2, N_NODES, K2P);
  k_bn_coeff<<<1, HIDC, 0, stream>>>(stats2, bn2_g, bn2_b, 1.0f / N_NODES, HIDC, coeff2);
  k_bn_to_xcat<<<gdiv((long long)N_NODES * HIDC, 256), 256, 0, stream>>>(y2, coeff2, xcat, HIDC);

  // MLP on concat [x1|x2] + BN + per-graph max pool
  k_gemm_wmma<16><<<N_NODES / 64, 128, 0, stream>>>(xcat, W3p, mlp_b, y3, stats3, N_NODES, K3P);
  k_bn_coeff<<<1, 256, 0, stream>>>(stats3, mlp_g, mlp_be, 1.0f / N_NODES, 256, coeff3);
  k_bn_max<<<gdiv((long long)N_NODES * 256, 256), 256, 0, stream>>>(y3, coeff3, batch, gmax);

  // head
  k_head<<<1, 256, 0, stream>>>(gmax, f1w, f1b, f1g, f1be, f2w, f2b, f2g, f2be,
                                f3w, f3b, (float*)d_out);
}
